// InteractionBlock_22643067585250
// MI455X (gfx1250) — matrix operand order repaired and checked by reading, not compiled, not run on hardware
//
#include <hip/hip_runtime.h>
#include <math.h>

// ---------------------------------------------------------------------------
// InteractionBlock (ViT-Adapter injector/extractor + MSDeformAttn) for MI455X
// Dense projections: v_wmma_f32_16x16x32_bf16 (wave32). GEMM staging uses
// CDNA5 GLOBAL_LOAD_ASYNC_TO_LDS_B128 (ASYNCcnt) into double-buffered LDS:
// tile k+1 streams global->LDS with zero VGPR traffic while tile k's WMMAs
// run; s_wait_asynccnt 0 + one barrier per K-step.
// ---------------------------------------------------------------------------

typedef __attribute__((ext_vector_type(16))) __bf16        v16bf;
typedef __attribute__((ext_vector_type(8)))  float         v8f;
typedef __attribute__((ext_vector_type(8)))  unsigned int  v8u;

#define DIMC   768
#define NHEADS 6
#define NPTS   4
#define HD     128           // DIMC / NHEADS
#define BB     4
#define NC     5376          // multiscale tokens per batch
#define NX     1024          // ViT tokens per batch

// ------------------------------ helpers ------------------------------------

__device__ __forceinline__ unsigned short f2bf(float f) {
  unsigned int u = __float_as_uint(f);
  u += 0x7FFFu + ((u >> 16) & 1u);          // round-to-nearest-even
  return (unsigned short)(u >> 16);
}

__device__ __forceinline__ float gelu_exact(float x) {
  return 0.5f * x * (1.0f + erff(x * 0.7071067811865476f));
}

// ------------------------------ WMMA GEMM ----------------------------------
// C[M,N] = A[M,K] * W[N,K]^T + bias[N]   (bf16 in, f32 accumulate/out)
// Block: 128 threads = 4 waves; block tile 128x64; wave tile 32x64.
#define GTM 128
#define GTN 64
#define GTK 32
#define LDA 40               // padded LDS row stride (ushorts); 80B, 16B-mult
#define ABUF (GTM * LDA * 2) // bytes per A buffer (10240)
#define BBUF (GTN * LDA * 2) // bytes per B buffer (5120)

__global__ __launch_bounds__(128, 1) void k_gemm_bf16(
    const unsigned short* __restrict__ A,   // [M,K] bf16 bits
    const unsigned short* __restrict__ W,   // [N,K] bf16 bits
    const float* __restrict__ bias, float* __restrict__ C,
    int M, int N, int K)
{
  __shared__ unsigned short lA[2][GTM * LDA];  // 2 x 10KB
  __shared__ unsigned short lB[2][GTN * LDA];  // 2 x  5KB

  const int tid  = threadIdx.x;
  const int lane = tid & 31;
  const int wave = tid >> 5;
  const int rowBase = blockIdx.y * GTM;
  const int colBase = blockIdx.x * GTN;
  const int r = lane & 15;
  const int h = lane >> 4;
  const int wRow = wave * 32;

  // ---- staging addresses (index-clamped: OOB rows/cols only ever feed
  //      output tiles the epilogue never stores, so no zero-fill needed)
  const unsigned short* pA[4];
  unsigned ldsA[4];
#pragma unroll
  for (int j = 0; j < 4; ++j) {
    int i = tid + 128 * j;                  // 512 chunks of 8 ushorts
    int m = i >> 2, k8 = (i & 3) << 3;
    int gm = rowBase + m; if (gm > M - 1) gm = M - 1;
    pA[j] = A + (size_t)gm * K + k8;
    ldsA[j] = (unsigned)(size_t)&lA[0][m * LDA + k8];   // LDS byte offset
  }
  const unsigned short* pB[2];
  unsigned ldsB[2];
#pragma unroll
  for (int j = 0; j < 2; ++j) {
    int i = tid + 128 * j;                  // 256 chunks of 8 ushorts
    int n = i >> 2, k8 = (i & 3) << 3;
    int gn = colBase + n; if (gn > N - 1) gn = N - 1;
    pB[j] = W + (size_t)gn * K + k8;
    ldsB[j] = (unsigned)(size_t)&lB[0][n * LDA + k8];
  }

  // CDNA5 async tensor path: global -> LDS with no VGPR data movement
  auto asyncTile = [&](int k0, int buf) {
    const unsigned offA = (unsigned)(buf * ABUF);
    const unsigned offB = (unsigned)(buf * BBUF);
#pragma unroll
    for (int j = 0; j < 4; ++j)
      asm volatile("global_load_async_to_lds_b128 %0, %1, off"
                   :: "v"(ldsA[j] + offA), "v"(pA[j] + k0) : "memory");
#pragma unroll
    for (int j = 0; j < 2; ++j)
      asm volatile("global_load_async_to_lds_b128 %0, %1, off"
                   :: "v"(ldsB[j] + offB), "v"(pB[j] + k0) : "memory");
  };

  v8f zf = {};
  v8f acc[2][4];
#pragma unroll
  for (int i = 0; i < 2; ++i)
#pragma unroll
    for (int j = 0; j < 4; ++j) acc[i][j] = zf;

  asyncTile(0, 0);
  asm volatile("s_wait_asynccnt 0x0" ::: "memory");
  __syncthreads();

  int cur = 0;
  for (int k0 = 0; k0 < K; k0 += GTK) {
    const bool notLast = (k0 + GTK < K);
    if (notLast) asyncTile(k0 + GTK, cur ^ 1);  // stream next tile during WMMAs
    if (k0 + 2 * GTK < K)                       // L2 prefetch two tiles ahead
      __builtin_prefetch(pA[0] + k0 + 2 * GTK, 0, 1);

    const unsigned short* bufA = lA[cur];
    const unsigned short* bufB = lB[cur];

    // ---- fragments (ISA 7.12.2 wave32 layouts)
    // A 16x32: lane r = row, half h; VGPR v pair at ka = 2*(v&3)+8h+16*(v>>2)
    // B 32x16: lane r = col, half h; VGPR v pair at kb = 2*(v&3)+8h+16*(v>>2)
    v8u pa[2], pb[4];
#pragma unroll
    for (int v = 0; v < 8; ++v) {
      const int ka = 2 * (v & 3) + 8 * h + 16 * (v >> 2);
      const int kb = 2 * (v & 3) + 8 * h + 16 * (v >> 2);
      pa[0][v] = *(const unsigned int*)&bufA[(wRow +      r) * LDA + ka];
      pa[1][v] = *(const unsigned int*)&bufA[(wRow + 16 + r) * LDA + ka];
      pb[0][v] = *(const unsigned int*)&bufB[(      r) * LDA + kb];
      pb[1][v] = *(const unsigned int*)&bufB[(16 + r) * LDA + kb];
      pb[2][v] = *(const unsigned int*)&bufB[(32 + r) * LDA + kb];
      pb[3][v] = *(const unsigned int*)&bufB[(48 + r) * LDA + kb];
    }
    v16bf a0 = __builtin_bit_cast(v16bf, pa[0]);
    v16bf a1 = __builtin_bit_cast(v16bf, pa[1]);
    v16bf b0 = __builtin_bit_cast(v16bf, pb[0]);
    v16bf b1 = __builtin_bit_cast(v16bf, pb[1]);
    v16bf b2 = __builtin_bit_cast(v16bf, pb[2]);
    v16bf b3 = __builtin_bit_cast(v16bf, pb[3]);

    acc[0][0] = __builtin_amdgcn_wmma_f32_16x16x32_bf16(false, a0, false, b0, (short)0, acc[0][0], false, false);
    acc[0][1] = __builtin_amdgcn_wmma_f32_16x16x32_bf16(false, a0, false, b1, (short)0, acc[0][1], false, false);
    acc[0][2] = __builtin_amdgcn_wmma_f32_16x16x32_bf16(false, a0, false, b2, (short)0, acc[0][2], false, false);
    acc[0][3] = __builtin_amdgcn_wmma_f32_16x16x32_bf16(false, a0, false, b3, (short)0, acc[0][3], false, false);
    acc[1][0] = __builtin_amdgcn_wmma_f32_16x16x32_bf16(false, a1, false, b0, (short)0, acc[1][0], false, false);
    acc[1][1] = __builtin_amdgcn_wmma_f32_16x16x32_bf16(false, a1, false, b1, (short)0, acc[1][1], false, false);
    acc[1][2] = __builtin_amdgcn_wmma_f32_16x16x32_bf16(false, a1, false, b2, (short)0, acc[1][2], false, false);
    acc[1][3] = __builtin_amdgcn_wmma_f32_16x16x32_bf16(false, a1, false, b3, (short)0, acc[1][3], false, false);

    if (notLast) {
      // own fragment reads retired (DScnt waited before WMMAs), async writes
      // of next tile land in the other buffer: wait + one barrier
      asm volatile("s_wait_asynccnt 0x0" ::: "memory");
      __syncthreads();
    }
    cur ^= 1;
  }

  // ---- epilogue: C/D layout VGPR i -> row = i + 8h, col = lane&15
  if (rowBase + GTM <= M && colBase + GTN <= N) {       // full-tile fast path
#pragma unroll
    for (int j = 0; j < 4; ++j) {
      int gc = colBase + j * 16 + r;
      float bv = bias[gc];
#pragma unroll
      for (int i2 = 0; i2 < 2; ++i2) {
        size_t base = (size_t)(rowBase + wRow + i2 * 16 + 8 * h) * N + gc;
#pragma unroll
        for (int i = 0; i < 8; ++i)
          C[base + (size_t)i * N] = acc[i2][j][i] + bv;
      }
    }
  } else {
#pragma unroll
    for (int j = 0; j < 4; ++j) {
      int gc = colBase + j * 16 + r;
      float bv = (gc < N) ? bias[gc] : 0.0f;
#pragma unroll
      for (int i2 = 0; i2 < 2; ++i2) {
        int rb0 = rowBase + wRow + i2 * 16 + 8 * h;
#pragma unroll
        for (int i = 0; i < 8; ++i) {
          int gr = rb0 + i;
          if (gr < M && gc < N) C[(size_t)gr * N + gc] = acc[i2][j][i] + bv;
        }
      }
    }
  }
}

// ------------------------------ weight convert -----------------------------
__global__ void k_cvt_bf16(const float* __restrict__ in,
                           unsigned short* __restrict__ out, unsigned n)
{
  unsigned i = blockIdx.x * 256 + threadIdx.x;
  if (i < n) out[i] = f2bf(in[i]);
}

// ------------------------------ LayerNorm ----------------------------------
// one wave per row, shuffle reduction (wave32); bf16 output feeds GEMMs
__global__ __launch_bounds__(256) void k_layernorm(
    const float* __restrict__ in, const float* __restrict__ g,
    const float* __restrict__ b, unsigned short* __restrict__ out, int M)
{
  int row = blockIdx.x * 8 + (threadIdx.x >> 5);
  if (row >= M) return;
  int lane = threadIdx.x & 31;
  const float* rp = in + (size_t)row * DIMC;
  float s = 0.0f, s2 = 0.0f;
  for (int ci = lane; ci < DIMC; ci += 32) { float v = rp[ci]; s += v; s2 += v * v; }
#pragma unroll
  for (int o = 16; o > 0; o >>= 1) {
    s  += __shfl_xor(s,  o, 32);
    s2 += __shfl_xor(s2, o, 32);
  }
  float mean = s * (1.0f / DIMC);
  float var  = fmaxf(s2 * (1.0f / DIMC) - mean * mean, 0.0f);
  float inv  = rsqrtf(var + 1e-6f);
  unsigned short* op = out + (size_t)row * DIMC;
  for (int ci = lane; ci < DIMC; ci += 32)
    op[ci] = f2bf((rp[ci] - mean) * inv * g[ci] + b[ci]);
}

// ------------------------------ elementwise --------------------------------
// HW is a power of two (4096/1024/256): hwShift = log2(HW)

__global__ void k_gather(const float* __restrict__ src,
                         unsigned short* __restrict__ dst,
                         unsigned total, int hwShift, int srcOff)
{
  unsigned i = blockIdx.x * 256 + threadIdx.x;
  if (i >= total) return;
  unsigned ch  = i % DIMC;
  unsigned row = i / DIMC;                    // b*HW + t
  unsigned t   = row & ((1u << hwShift) - 1u);
  unsigned bb  = row >> hwShift;
  dst[i] = f2bf(src[((size_t)bb * NC + srcOff + t) * DIMC + ch]);
}

__global__ void k_scatter_add(const float* __restrict__ base,
                              const float* __restrict__ add,
                              float* __restrict__ out,
                              unsigned total, int hwShift, int dstOff)
{
  unsigned i = blockIdx.x * 256 + threadIdx.x;
  if (i >= total) return;
  unsigned ch  = i % DIMC;
  unsigned row = i / DIMC;
  unsigned t   = row & ((1u << hwShift) - 1u);
  unsigned bb  = row >> hwShift;
  size_t di = ((size_t)bb * NC + dstOff + t) * DIMC + ch;
  out[di] = base[di] + add[i];
}

__global__ void k_add(const float* __restrict__ a, const float* __restrict__ b,
                      float* __restrict__ out, unsigned total)
{
  unsigned i = blockIdx.x * 256 + threadIdx.x;
  if (i < total) out[i] = a[i] + b[i];
}

__global__ void k_add_inplace(float* __restrict__ out, const float* __restrict__ a,
                              unsigned total)
{
  unsigned i = blockIdx.x * 256 + threadIdx.x;
  if (i < total) out[i] += a[i];
}

__global__ void k_axpy_gamma(const float* __restrict__ x, const float* __restrict__ gamma,
                             const float* __restrict__ a, float* __restrict__ out,
                             unsigned total)
{
  unsigned i = blockIdx.x * 256 + threadIdx.x;
  if (i < total) out[i] = x[i] + gamma[i % DIMC] * a[i];
}

// ------------------------------ ConvGLU mid --------------------------------
// h: [B*HW, 512] f32; u = h[:, :256] -> dwconv kxk -> gelu, gated by h[:,256:]
__global__ void k_cglu_mid(const float* __restrict__ h, const float* __restrict__ dww,
                           const float* __restrict__ dwb,
                           unsigned short* __restrict__ out,
                           unsigned total, int hwShift, int wShift, int ksz)
{
  unsigned i = blockIdx.x * 256 + threadIdx.x;
  if (i >= total) return;
  const int Ws = 1 << wShift, Hs = 1 << (hwShift - wShift);
  unsigned ch  = i & 255u;
  unsigned row = i >> 8;                       // b*HW + t
  unsigned t   = row & ((1u << hwShift) - 1u);
  unsigned bb  = row >> hwShift;
  int y = (int)(t >> wShift), x = (int)(t & (unsigned)(Ws - 1));
  int p = ksz >> 1;
  float s = dwb[ch];
  for (int ky = 0; ky < ksz; ++ky) {
    int iy = y + ky - p;
    if (iy < 0 || iy >= Hs) continue;
    for (int kx = 0; kx < ksz; ++kx) {
      int ix = x + kx - p;
      if (ix < 0 || ix >= Ws) continue;
      s += h[(((size_t)bb << hwShift) + (iy << wShift) + ix) * 512 + ch] *
           dww[(ch * ksz + ky) * ksz + kx];
    }
  }
  float u = gelu_exact(s);
  float v = h[(size_t)row * 512 + 256 + ch];
  out[(size_t)row * 256 + ch] = f2bf(u * v);
}

// ------------------------------ ConvFFN mid --------------------------------
// shared 3x3 depthwise over the 3 scales of the 5376-token pyramid, then GELU
__global__ void k_ffn_mid(const float* __restrict__ h, const float* __restrict__ dww,
                          const float* __restrict__ dwb,
                          unsigned short* __restrict__ out)
{
  unsigned i = blockIdx.x * 256 + threadIdx.x;
  const unsigned total = (unsigned)BB * NC * 192;
  if (i >= total) return;
  unsigned ch = i % 192u;
  unsigned r  = i / 192u;
  unsigned t  = r % (unsigned)NC;
  unsigned bb = r / (unsigned)NC;
  int Hs, Ws, off;
  if (t < 4096u)      { Hs = 64; Ws = 64; off = 0; }
  else if (t < 5120u) { Hs = 32; Ws = 32; off = 4096; }
  else                { Hs = 16; Ws = 16; off = 5120; }
  int tt = (int)t - off, y = tt / Ws, x = tt % Ws;
  float s = dwb[ch];
  for (int ky = 0; ky < 3; ++ky) {
    int iy = y + ky - 1;
    if (iy < 0 || iy >= Hs) continue;
    for (int kx = 0; kx < 3; ++kx) {
      int ix = x + kx - 1;
      if (ix < 0 || ix >= Ws) continue;
      s += h[((size_t)bb * NC + off + iy * Ws + ix) * 192 + ch] *
           dww[(ch * 3 + ky) * 3 + kx];
    }
  }
  out[(size_t)r * 192 + ch] = f2bf(gelu_exact(s));
}

// ------------------------------ softmax over L*P ---------------------------
__global__ void k_softmax_lp(float* __restrict__ aw, unsigned M, int LP)
{
  unsigned i = blockIdx.x * 256 + threadIdx.x;
  unsigned total = M * NHEADS;
  if (i >= total) return;
  unsigned hd = i % NHEADS, row = i / NHEADS;
  float* p = aw + (size_t)row * NHEADS * LP + hd * LP;
  float mx = p[0];
  for (int j = 1; j < LP; ++j) mx = fmaxf(mx, p[j]);
  float s = 0.0f;
  for (int j = 0; j < LP; ++j) { float e = __expf(p[j] - mx); p[j] = e; s += e; }
  float inv = 1.0f / s;
  for (int j = 0; j < LP; ++j) p[j] *= inv;
}

// ------------------------------ deformable sampling ------------------------
// one 128-thread block per (b, q, head); thread d accumulates channel d.
__global__ __launch_bounds__(128) void k_msda_sample(
    const float* __restrict__ val,           // [B*S, 768] f32
    const float* __restrict__ ref,           // [B, Lq, nL, 2]
    const float* __restrict__ off,           // [B*Lq, NHEADS*nL*NPTS*2]
    const float* __restrict__ aw,            // [B*Lq, NHEADS*nL*NPTS]
    unsigned short* __restrict__ out,        // [B*Lq, 768] bf16
    int Lq, int S, int nL,
    int H0, int W0, int O0, int H1, int W1, int O1, int H2, int W2, int O2)
{
  int blk = blockIdx.x;
  int hd = blk % NHEADS;
  int q  = (blk / NHEADS) % Lq;
  int bb = blk / (NHEADS * Lq);
  int d  = threadIdx.x;

  const float* refp = ref + (((size_t)bb * Lq + q) * nL) * 2;
  const float* offp = off + ((size_t)bb * Lq + q) * (NHEADS * nL * NPTS * 2)
                          + hd * nL * NPTS * 2;
  const float* awp  = aw  + ((size_t)bb * Lq + q) * (NHEADS * nL * NPTS)
                          + hd * nL * NPTS;
  const size_t chan = (size_t)hd * HD + d;

  float acc = 0.0f;
  for (int l = 0; l < nL; ++l) {
    const int Hl = (l == 0) ? H0 : ((l == 1) ? H1 : H2);
    const int Wl = (l == 0) ? W0 : ((l == 1) ? W1 : W2);
    const int Ol = (l == 0) ? O0 : ((l == 1) ? O1 : O2);
    const float rx = refp[l * 2 + 0];
    const float ry = refp[l * 2 + 1];
    const float* vbase = val + ((size_t)bb * S + Ol) * DIMC + chan;
#pragma unroll
    for (int p = 0; p < NPTS; ++p) {
      float ox = offp[(l * NPTS + p) * 2 + 0];
      float oy = offp[(l * NPTS + p) * 2 + 1];
      float w  = awp[l * NPTS + p];
      // loc = ref + off/[W,H];  x = loc_x*W - 0.5 = ref_x*W + off_x - 0.5
      float xx = rx * (float)Wl + ox - 0.5f;
      float yy = ry * (float)Hl + oy - 0.5f;
      float xf = floorf(xx), yf = floorf(yy);
      int x0 = (int)xf, y0 = (int)yf;
      float wx = xx - xf, wy = yy - yf;
      float w00 = (1.0f - wy) * (1.0f - wx);
      float w01 = (1.0f - wy) * wx;
      float w10 = wy * (1.0f - wx);
      float w11 = wy * wx;
      float s = 0.0f;
      if (y0 >= 0 && y0 < Hl) {
        if (x0 >= 0 && x0 < Wl)
          s += w00 * vbase[(size_t)(y0 * Wl + x0) * DIMC];
        if (x0 + 1 >= 0 && x0 + 1 < Wl)
          s += w01 * vbase[(size_t)(y0 * Wl + x0 + 1) * DIMC];
      }
      if (y0 + 1 >= 0 && y0 + 1 < Hl) {
        if (x0 >= 0 && x0 < Wl)
          s += w10 * vbase[(size_t)((y0 + 1) * Wl + x0) * DIMC];
        if (x0 + 1 >= 0 && x0 + 1 < Wl)
          s += w11 * vbase[(size_t)((y0 + 1) * Wl + x0 + 1) * DIMC];
      }
      acc += w * s;
    }
  }
  out[((size_t)bb * Lq + q) * DIMC + chan] = f2bf(acc);
}

// ------------------------------ host orchestration -------------------------

// param indices in d_in after pytree flatten (dict keys sorted, jax-style)
enum {
  IN_X = 0, IN_C, IN_REF1, IN_REF2,
  E_AW_B = 4, E_AW_W, E_FFN_DW_B, E_FFN_DW_W, E_FFN_FC1_B, E_FFN_FC1_W,
  E_FFN_FC2_B, E_FFN_FC2_W, E_FFN_N_B, E_FFN_N_G, E_FN_B, E_FN_G,
  E_OFF_B, E_OFF_W, E_OUT_B, E_OUT_W, E_QN_B, E_QN_G, E_VAL_B, E_VAL_W,
  I_AW_B = 24, I_AW_W, I_FN_B, I_FN_G, I_GAMMA, I_OFF_B, I_OFF_W,
  I_OUT_B, I_OUT_W, I_QN_B, I_QN_G, I_VAL_B, I_VAL_W,
  M_BASE = 37     // cglu1 @37, cglu2 @43, cglu3 @49; each: dw_b,dw_w,fc1_b,fc1_w,fc2_b,fc2_w
};

static inline unsigned grid1(size_t total) {
  return (unsigned)((total + 255) / 256);
}

extern "C" void kernel_launch(void* const* d_in, const int* in_sizes, int n_in,
                              void* d_out, int out_size, void* d_ws, size_t ws_size,
                              hipStream_t stream) {
  (void)in_sizes; (void)n_in; (void)out_size; (void)ws_size;

  auto P = [&](int i) { return (const float*)d_in[i]; };

  const float* x    = P(IN_X);     // [4096, 768]
  const float* c    = P(IN_C);     // [21504, 768]
  const float* ref1 = P(IN_REF1);  // [4, 1024, 3, 2]
  const float* ref2 = P(IN_REF2);  // [4, 5376, 1, 2]

  const size_t MC = (size_t)BB * NC;   // 21504
  const size_t MX = (size_t)BB * NX;   // 4096
  const size_t RSZ = MC * DIMC;        // elements per region

  float* R0 = (float*)d_ws;            // c2 = c + msgm(c)          (f32)
  float* R1 = R0 + RSZ;                //                           (f32)
  float* R2 = R1 + RSZ;                //                           (f32)
  unsigned short* H1 = (unsigned short*)(R2 + RSZ);   // bf16 activations
  unsigned short* H2 = H1 + RSZ;                      // bf16 activations
  unsigned short* WB = H2 + RSZ;                      // bf16 weights (<=768x768)

  float* outX = (float*)d_out;                 // xq out  [4096, 768]
  float* outC = outX + MX * DIMC;              // c  out  [21504, 768]

  // convert weight + run WMMA GEMM
  auto gemmw = [&](const unsigned short* Abf, int widx, int bidx, float* Cout,
                   int M, int N, int K) {
    k_cvt_bf16<<<grid1((size_t)N * K), 256, 0, stream>>>(P(widx), WB, (unsigned)(N * K));
    dim3 g((N + GTN - 1) / GTN, (M + GTM - 1) / GTM);
    k_gemm_bf16<<<g, dim3(128), 0, stream>>>(Abf, WB, P(bidx), Cout, M, N, K);
  };
  auto ln = [&](const float* in, const float* g, const float* b,
                unsigned short* out, int M) {
    k_layernorm<<<dim3((M + 7) / 8), dim3(256), 0, stream>>>(in, g, b, out, M);
  };

  // ---------------- Phase M: c2 = c + msgm(c) ----------------
  const int scHW[3]   = {4096, 1024, 256};
  const int scHWS[3]  = {12, 10, 8};     // log2(HW)
  const int scWS[3]   = {6, 5, 4};       // log2(W)
  const int scOff[3]  = {0, 4096, 5120};
  const int scK[3]    = {5, 3, 3};
  for (int s = 0; s < 3; ++s) {
    const int pi = M_BASE + 6 * s;   // dw_b, dw_w, fc1_b, fc1_w, fc2_b, fc2_w
    const int Mrows = BB * scHW[s];
    const unsigned tG = (unsigned)Mrows * DIMC;
    const unsigned tM = (unsigned)Mrows * 256;
    k_gather<<<grid1(tG), 256, 0, stream>>>(c, H1, tG, scHWS[s], scOff[s]);
    gemmw(H1, pi + 3, pi + 2, R1, Mrows, 512, DIMC);                   // fc1
    k_cglu_mid<<<grid1(tM), 256, 0, stream>>>(
        R1, P(pi + 1), P(pi + 0), H2, tM, scHWS[s], scWS[s], scK[s]);
    gemmw(H2, pi + 5, pi + 4, R2, Mrows, DIMC, 256);                   // fc2
    k_scatter_add<<<grid1(tG), 256, 0, stream>>>(
        c, R2, R0, tG, scHWS[s], scOff[s]);                            // c2 rows
  }

  // ---------------- Phase I: injector (x queries c2) ----------------
  ln(x,  P(I_QN_G), P(I_QN_B), H1, (int)MX);                 // Q  (bf16)
  ln(R0, P(I_FN_G), P(I_FN_B), H2, (int)MC);                 // Fn (bf16)
  gemmw(H2, I_VAL_W, I_VAL_B, R1, (int)MC, DIMC, DIMC);      // V [21504,768]
  float* OFFi = R2;
  float* AWi  = R2 + MX * 144;
  gemmw(H1, I_OFF_W, I_OFF_B, OFFi, (int)MX, 144, DIMC);
  gemmw(H1, I_AW_W,  I_AW_B,  AWi,  (int)MX, 72,  DIMC);
  k_softmax_lp<<<grid1(MX * NHEADS), 256, 0, stream>>>(AWi, (unsigned)MX, 12);
  k_msda_sample<<<dim3((unsigned)(MX * NHEADS)), dim3(HD), 0, stream>>>(
      R1, ref1, OFFi, AWi, H1, NX, NC, 3,
      64, 64, 0, 32, 32, 4096, 16, 16, 5120);                // S -> H1 (Q dead)
  gemmw(H1, I_OUT_W, I_OUT_B, R2, (int)MX, DIMC, DIMC);      // attn -> R2
  k_axpy_gamma<<<grid1(MX * DIMC), 256, 0, stream>>>(
      x, P(I_GAMMA), R2, outX, (unsigned)(MX * DIMC));       // xq out

  // ---------------- Phase E: extractor (c2 queries xq) ----------------
  ln(R0,   P(E_QN_G), P(E_QN_B), H1, (int)MC);               // Qc (bf16)
  ln(outX, P(E_FN_G), P(E_FN_B), H2, (int)MX);               // Fx (bf16)
  gemmw(H2, E_VAL_W, E_VAL_B, R1, (int)MX, DIMC, DIMC);      // Vx [4096,768]
  float* OFFe = R2;
  float* AWe  = R2 + MC * 48;
  gemmw(H1, E_OFF_W, E_OFF_B, OFFe, (int)MC, 48, DIMC);
  gemmw(H1, E_AW_W,  E_AW_B,  AWe,  (int)MC, 24, DIMC);
  k_softmax_lp<<<grid1(MC * NHEADS), 256, 0, stream>>>(AWe, (unsigned)MC, 4);
  k_msda_sample<<<dim3((unsigned)(MC * NHEADS)), dim3(HD), 0, stream>>>(
      R1, ref2, OFFe, AWe, H2, NC, NX, 1,
      32, 32, 0, 1, 1, 0, 1, 1, 0);                          // Se -> H2 (Fx dead)
  gemmw(H2, E_OUT_W, E_OUT_B, R1, (int)MC, DIMC, DIMC);      // attn -> R1
  k_add<<<grid1(MC * DIMC), 256, 0, stream>>>(
      R0, R1, outC, (unsigned)(MC * DIMC));                  // c3
  // ---------------- Phase F: ConvFFN on c3 ----------------
  ln(outC, P(E_FFN_N_G), P(E_FFN_N_B), H1, (int)MC);
  gemmw(H1, E_FFN_FC1_W, E_FFN_FC1_B, R1, (int)MC, 192, DIMC);
  k_ffn_mid<<<grid1(MC * 192), 256, 0, stream>>>(R1, P(E_FFN_DW_W), P(E_FFN_DW_B), H2);
  gemmw(H2, E_FFN_FC2_W, E_FFN_FC2_B, R2, (int)MC, DIMC, 192);
  k_add_inplace<<<grid1(MC * DIMC), 256, 0, stream>>>(
      outC, R2, (unsigned)(MC * DIMC));
}
